// TraceSeq2Seq_16054587752963
// MI455X (gfx1250) — matrix operand order repaired and checked by reading, not compiled, not run on hardware
//
#include <hip/hip_runtime.h>
#include <hip/hip_bf16.h>

typedef unsigned short ushort_t;
typedef __attribute__((ext_vector_type(16))) __bf16 v16bf;
typedef __attribute__((ext_vector_type(8)))  float  v8f;
typedef __attribute__((ext_vector_type(4)))  unsigned int uint4v;

union Frag { uint4v q[2]; v16bf v; };

#define BSZ   1024
#define SEQL  200
#define KX    192      // padded input-feature K (175 -> 192; 172..174 dynamic, 175..191 zero)
#define KH    256      // hidden
#define NG    1024     // 4*RNN_H gates
#define KT_X  6        // 192/32
#define KT_H  8        // 256/32
#define KT_ALL 14
#define NO    96       // padded output cols
#define OUTD  90

// ---- bf16 helpers (round-to-nearest-even) ----
__device__ __forceinline__ ushort_t f2bf(float f) {
  unsigned u = __float_as_uint(f);
  unsigned r = u + 0x7FFFu + ((u >> 16) & 1u);
  return (ushort_t)(r >> 16);
}
__device__ __forceinline__ float bf2f(ushort_t b) {
  return __uint_as_float(((unsigned)b) << 16);
}

// ---- fast transcendentals on the gfx1250 TRANS32 path ----
__device__ __forceinline__ float fast_rcp(float x) {
#if __has_builtin(__builtin_amdgcn_rcpf)
  return __builtin_amdgcn_rcpf(x);
#else
  return 1.f / x;
#endif
}
__device__ __forceinline__ float fast_tanh(float x) {
#if __has_builtin(__builtin_amdgcn_tanhf)
  return __builtin_amdgcn_tanhf(x);           // v_tanh_f32 (new in gfx1250)
#else
  // branch-free: tanh(x) = 1 - 2/(exp(2x)+1), stays on v_exp_f32
  float e = __expf(2.f * x);
  return 1.f - 2.f * fast_rcp(e + 1.f);
#endif
}
__device__ __forceinline__ float fast_sigmoid(float x) {
  return fast_rcp(1.f + __expf(-x));          // v_exp_f32 + v_rcp_f32
}

// =====================================================================
// Kernel 1: pack weights into bf16 WMMA-B fragment-contiguous layout.
// Wg[kt][n][32] : kt 0..5  -> W_ih rows kt*32+j (0 for k>=175)
//                 kt 6..13 -> W_hh rows (kt-6)*32+j
// Wo[kt][n][32] : W_out, n padded 90->96
// biasG[n] = b_ih[n] + b_hh[n]
// =====================================================================
__global__ __launch_bounds__(256) void prep_kernel(
    const float* __restrict__ W_ih, const float* __restrict__ b_ih,
    const float* __restrict__ W_hh, const float* __restrict__ b_hh,
    const float* __restrict__ W_out,
    ushort_t* __restrict__ Wg, ushort_t* __restrict__ Wo,
    float* __restrict__ biasG)
{
  int i = blockIdx.x * 256 + threadIdx.x;
  if (i < 14 * 1024 * 32) {
    int kt  = i >> 15;          // /(1024*32)
    int rem = i & 32767;
    int n   = rem >> 5;
    int j   = rem & 31;
    float v;
    if (kt < KT_X) { int k = kt * 32 + j; v = (k < 175) ? W_ih[k * 1024 + n] : 0.f; }
    else           { int k = (kt - KT_X) * 32 + j; v = W_hh[k * 1024 + n]; }
    Wg[i] = f2bf(v);
  } else if (i < 14 * 1024 * 32 + 8 * 96 * 32) {
    int m   = i - 14 * 1024 * 32;
    int kt  = m / (96 * 32);
    int rem = m % (96 * 32);
    int n   = rem >> 5;
    int j   = rem & 31;
    int k   = kt * 32 + j;
    float v = (n < OUTD) ? W_out[k * OUTD + n] : 0.f;
    Wo[m] = f2bf(v);
  } else if (i < 14 * 1024 * 32 + 8 * 96 * 32 + NG) {
    int n = i - (14 * 1024 * 32 + 8 * 96 * 32);
    biasG[n] = b_ih[n] + b_hh[n];
  }
}

// =====================================================================
// Kernel 2: time-parallel set encoder + input feature assembly (bf16).
// Xin row r = t*BSZ + b, cols [0..43] ptrace, [44..171] enc (172..191 unused)
// =====================================================================
__global__ __launch_bounds__(256) void enc_kernel(
    const float* __restrict__ coords,
    const float* __restrict__ We1, const float* __restrict__ be1,
    const float* __restrict__ We2, const float* __restrict__ be2,
    ushort_t* __restrict__ Xin)
{
  __shared__ float sW1[128], sB1[64], sW2[64 * 64], sB2[64];
  int tid = threadIdx.x;
  for (int i = tid; i < 128; i += 256) sW1[i] = We1[i];
  for (int i = tid; i < 64; i += 256) { sB1[i] = be1[i]; sB2[i] = be2[i]; }
  for (int i = tid; i < 4096; i += 256) sW2[i] = We2[i];
  __syncthreads();

  int pos = blockIdx.x * 256 + tid;   // exactly SEQL*BSZ threads
  int b = pos & (BSZ - 1);
  int t = pos >> 10;
  const float* cb = coords + ((size_t)b * SEQL + t) * 88;
  ushort_t* xr = Xin + (size_t)pos * KX;

  #pragma unroll
  for (int p = 0; p < 22; ++p) {
    xr[2 * p]     = f2bf(cb[4 * p]);
    xr[2 * p + 1] = f2bf(cb[4 * p + 1]);
  }

  for (int team = 0; team < 2; ++team) {
    float m[64];
    #pragma unroll
    for (int j = 0; j < 64; ++j) m[j] = 0.f;
    for (int p = 0; p < 11; ++p) {
      float x = cb[4 * (team * 11 + p)];
      float y = cb[4 * (team * 11 + p) + 1];
      #pragma unroll
      for (int j = 0; j < 64; ++j) {
        float h = fmaf(x, sW1[j], fmaf(y, sW1[64 + j], sB1[j]));
        m[j] += fmaxf(h, 0.f);
      }
    }
    #pragma unroll
    for (int j = 0; j < 64; ++j) m[j] *= (1.f / 11.f);
    for (int j = 0; j < 64; ++j) {
      float e = sB2[j];
      #pragma unroll
      for (int i2 = 0; i2 < 64; ++i2) e = fmaf(m[i2], sW2[i2 * 64 + j], e);
      xr[44 + team * 64 + j] = f2bf(e);
    }
  }
}

// =====================================================================
// Kernel 3: persistent LSTM scan. 64 blocks x 16 batch rows, 8 waves.
// Gate mapping: wave wv owns hidden slice [wv*32, wv*32+32) of ALL FOUR
// gates, so the LSTM nonlinearity runs in registers (cx register-resident).
// Per step: concurrent {X-tile load + dynamic-col patch + next-t prefetch},
// 14-ktile WMMA gate GEMM, in-register LSTM, 8-ktile WMMA output GEMM.
// =====================================================================
__global__ __launch_bounds__(256) void step_kernel(
    const float* __restrict__ coords,
    const float* __restrict__ pitch,
    const ushort_t* __restrict__ Wg,
    const ushort_t* __restrict__ Wo,
    const float* __restrict__ biasG,
    const float* __restrict__ b_out,
    const ushort_t* __restrict__ Xin,
    float* __restrict__ out)
{
  __shared__ __align__(16) ushort_t Xs[16 * KX];   // A-tile (input features)
  __shared__ __align__(16) ushort_t Hs[16 * KH];   // hx (bf16, A-layout row-major)
  __shared__ float Pred[32];                       // prev pred (x,y) per row
  __shared__ float BiasS[NG];
  __shared__ float Psc[2];

  const int tid    = threadIdx.x;
  const int lane   = tid & 31;          // wave32
  const int wv     = tid >> 5;          // 8 waves
  const int laneLo = lane & 15;
  const int laneHi = lane >> 4;
  const int b0     = blockIdx.x * 16;

  for (int i = tid; i < 16 * KH; i += 256) Hs[i] = 0;
  for (int i = tid; i < NG; i += 256) BiasS[i] = biasG[i];
  if (tid < 32) Pred[tid] = 0.f;
  if (tid < 2)  Psc[tid] = pitch[tid];
  for (int i = tid; i < 16 * 20; i += 256) {       // zero cols 172..191 once
    int row = i / 20, c = 172 + i % 20;
    Xs[row * KX + c] = 0;
  }
  __syncthreads();

  v8f cxr[2] = {};                                 // cell state, register-resident

  for (int t = 0; t < SEQL; ++t) {
    // ---- load live 16x172 cols of A-tile (16x86 dwords), patch dynamic
    //      cols from global coords + Pred concurrently (disjoint dwords),
    //      and prefetch next step's tile into L2/WGP$ ----
    #pragma unroll
    for (int q = 0; q < 6; ++q) {
      int idx = tid + q * 256;
      if (idx < 16 * 86) {
        int row = idx / 86, c = idx % 86;
        ((unsigned*)Xs)[row * 96 + c] =
            ((const unsigned*)(Xin + ((size_t)t * BSZ + b0 + row) * KX))[c];
      }
    }
    if (tid < 16) {
      const float* cb = coords + ((size_t)(b0 + tid) * SEQL + t) * 88;
      float px = Pred[tid * 2 + 0], py = Pred[tid * 2 + 1];
      float best = 3.4e38f;
      #pragma unroll
      for (int p = 0; p < 22; ++p) {
        float dx = px - cb[4 * p];
        float dy = py - cb[4 * p + 1];
        best = fminf(best, sqrtf(dx * dx + dy * dy));
      }
      Xs[tid * KX + 172] = f2bf(best);
      Xs[tid * KX + 173] = f2bf(px);
      Xs[tid * KX + 174] = f2bf(py);
    }
    if ((t + 1 < SEQL) && tid < 48) {              // 16 rows x 3 cachelines
      int row = tid / 3, seg = tid % 3;
      __builtin_prefetch(Xin + ((size_t)(t + 1) * BSZ + b0 + row) * KX + seg * 64, 0, 1);
    }
    __syncthreads();

    // ---- gate GEMM: tiles j -> col = (j>>1)*256 + wv*32 + (j&1)*16 + lane ----
    v8f acc[8] = {};
    {
      #pragma unroll 1
      for (int kt = 0; kt < KT_ALL; ++kt) {
        Frag a;
        if (kt < KT_X) {
          int kb = kt * 32 + laneHi * 8;
          a.q[0] = *(const uint4v*)&Xs[laneLo * KX + kb];
          a.q[1] = *(const uint4v*)&Xs[laneLo * KX + kb + 16];
        } else {
          int kb = (kt - KT_X) * 32 + laneHi * 8;
          a.q[0] = *(const uint4v*)&Hs[laneLo * KH + kb];
          a.q[1] = *(const uint4v*)&Hs[laneLo * KH + kb + 16];
        }
        #pragma unroll
        for (int j = 0; j < 8; ++j) {
          int ncol = (j >> 1) * 256 + wv * 32 + (j & 1) * 16 + laneLo;
          const ushort_t* bp = Wg + ((size_t)kt * NG + ncol) * 32 + laneHi * 8;
          Frag b;
          b.q[0] = *(const uint4v*)bp;
          b.q[1] = *(const uint4v*)(bp + 16);
          acc[j] = __builtin_amdgcn_wmma_f32_16x16x32_bf16(
              false, a.v, false, b.v, (short)0, acc[j], false, false);
        }
      }
    }
    __syncthreads();   // all waves done reading Hs before rewriting it

    // ---- LSTM nonlinearity, fully in registers; write hx2 -> Hs ----
    #pragma unroll
    for (int sub = 0; sub < 2; ++sub) {
      int h = wv * 32 + sub * 16 + laneLo;
      float bi = BiasS[h];
      float bfv = BiasS[256 + h];
      float bg = BiasS[512 + h];
      float bo_ = BiasS[768 + h];
      #pragma unroll
      for (int i = 0; i < 8; ++i) {
        float ig = acc[0 + sub][i] + bi;
        float fg = acc[2 + sub][i] + bfv;
        float gg = acc[4 + sub][i] + bg;
        float og = acc[6 + sub][i] + bo_;
        float c  = cxr[sub][i];
        float si = fast_sigmoid(ig);
        float sf = fast_sigmoid(fg);
        float so = fast_sigmoid(og);
        float c2 = sf * c + si * fast_tanh(gg);
        float h2 = so * fast_tanh(c2);
        cxr[sub][i] = c2;
        int row = i + laneHi * 8;
        Hs[row * KH + h] = f2bf(h2);
      }
    }
    __syncthreads();

    // ---- output GEMM: out[16x90] = hx2 @ W_out ; waves 0..5 ----
    if (wv < 6) {
      v8f ao = {};
      #pragma unroll 1
      for (int kt = 0; kt < KT_H; ++kt) {
        int kb = kt * 32 + laneHi * 8;
        Frag a;
        a.q[0] = *(const uint4v*)&Hs[laneLo * KH + kb];
        a.q[1] = *(const uint4v*)&Hs[laneLo * KH + kb + 16];
        int ncol = wv * 16 + laneLo;
        const ushort_t* bp = Wo + ((size_t)kt * NO + ncol) * 32 + laneHi * 8;
        Frag b;
        b.q[0] = *(const uint4v*)bp;
        b.q[1] = *(const uint4v*)(bp + 16);
        ao = __builtin_amdgcn_wmma_f32_16x16x32_bf16(
            false, a.v, false, b.v, (short)0, ao, false, false);
      }
      int col = wv * 16 + laneLo;
      if (col < OUTD) {
        float bo = b_out[col];
        float sc = Psc[col & 1];      // scale = tile(pitch_size,45)
        #pragma unroll
        for (int i = 0; i < 8; ++i) {
          int row = i + laneHi * 8;
          float val = (ao[i] + bo) * sc;
          out[((size_t)(b0 + row) * SEQL + t) * OUTD + col] = val;
          if (col == 88) Pred[row * 2 + 0] = val;
          if (col == 89) Pred[row * 2 + 1] = val;
        }
      }
    }
    __syncthreads();
  }
}

// =====================================================================
extern "C" void kernel_launch(void* const* d_in, const int* in_sizes, int n_in,
                              void* d_out, int out_size, void* d_ws, size_t ws_size,
                              hipStream_t stream) {
  (void)in_sizes; (void)n_in; (void)out_size; (void)ws_size;
  const float* coords = (const float*)d_in[0];
  const float* pitch  = (const float*)d_in[1];
  const float* W_e1   = (const float*)d_in[2];
  const float* b_e1   = (const float*)d_in[3];
  const float* W_e2   = (const float*)d_in[4];
  const float* b_e2   = (const float*)d_in[5];
  const float* W_ih   = (const float*)d_in[6];
  const float* b_ih   = (const float*)d_in[7];
  const float* W_hh   = (const float*)d_in[8];
  const float* b_hh   = (const float*)d_in[9];
  const float* W_out  = (const float*)d_in[10];
  const float* b_out  = (const float*)d_in[11];

  // workspace layout (bytes): Wg 917504 | Wo 49152 | biasG 4096 | Xin 78643200
  char* ws = (char*)d_ws;
  ushort_t* Wg    = (ushort_t*)(ws);
  ushort_t* Wo    = (ushort_t*)(ws + 917504);
  float*    biasG = (float*)   (ws + 966656);
  ushort_t* Xin   = (ushort_t*)(ws + 970752);

  prep_kernel<<<1892, 256, 0, stream>>>(W_ih, b_ih, W_hh, b_hh, W_out, Wg, Wo, biasG);
  enc_kernel<<<800, 256, 0, stream>>>(coords, W_e1, b_e1, W_e2, b_e2, Xin);
  step_kernel<<<64, 256, 0, stream>>>(coords, pitch, Wg, Wo, biasG, b_out, Xin,
                                      (float*)d_out);
}
